// Correlation_45286135169404
// MI455X (gfx1250) — matrix operand order repaired and checked by reading, not compiled, not run on hardware
//
#include <hip/hip_runtime.h>
#include <hip/hip_bf16.h>

// Correlation layer (FlowNet-style), MD=4 -> 81 displacements.
// B=16, C=256, H=W=96. out[b, (di+4)*9+(dj+4), i, j] =
//   (1/C) * sum_c in1[b,c,i,j] * in2[b,c,i+di,j+dj]   (zero-padded).
//
// Pass 1: f32 -> bf16 channel-pair packing into d_ws.
//         pk1[b][c/2][96][96]          (in1, unpadded; A tiles never OOB)
//         pk2[b][c/2][104][104]        (in2, zero 4-border; B fills guard-free)
// Pass 2: per (b, 4-row i tile, 16-wide j tile): the 16x9 (pixels x dj)
//         outputs for fixed di are a band of the 16x24 matrix A*B with
//         A = in1 row tile (16xC), B = in2 cols [j0-4, j0+19]. Two
//         v_wmma_f32_16x16x32_bf16 per 32-channel chunk cover the band;
//         9 waves/block (one per di) x 4 i-rows -> 8 WMMA/chunk/wave.
//         Double-buffered LDS, filled by async global->LDS copies (ASYNCcnt):
//         chunk kc+1 streams in while chunk kc runs on the WMMA pipe, with
//         per-wave partial s_wait_asynccnt (async loads complete in order).
// Fallback (small ws): single-pass kernel with in-kernel conversion.

typedef __attribute__((ext_vector_type(16))) __bf16        v16bf;
typedef __attribute__((ext_vector_type(8)))  float         v8f;
typedef __attribute__((ext_vector_type(8)))  unsigned int  v8u;
typedef __attribute__((ext_vector_type(4)))  unsigned int  v4u;

#define MD   4
#define NB   16
#define NC   256
#define NH   96
#define NW   96
#define ND   9              // number of di values
#define TJ   16             // j tile width (WMMA M)
#define TI   4              // i rows per block (main kernel)
#define KC   32             // K chunk (WMMA K for bf16)
#define NKC  (NC / KC)
#define KP   (KC / 2)       // k-pairs per chunk = 16
#define KPP  20             // padded k-pair row stride (16B-aligned rows,
                            // lane stride 20 dwords -> conflict-free banks)
#define BCOL 24             // B region columns: j0-4 .. j0+19
#define BROW (TI + 2 * MD)  // 12 rows: i0-4 .. i0+TI+3
#define NTHREADS (ND * 32)
#define HW   (NH * NW)      // 9216
#define PW   (NW + 2 * MD)  // 104
#define PH   (NH + 2 * MD)  // 104
#define PHW  (PW * PH)      // 10816
#define PKC  (NC / 2)       // 128 packed channel-pairs
#define ASZ  (TI * TJ * KPP)       // 1280 dwords per A buffer
#define BSZ  (BROW * BCOL * KPP)   // 5760 dwords per B buffer

// ---------------------------------------------------------------------------
// Async global->LDS copy (CDNA5 GLOBAL_LOAD_ASYNC_TO_LDS_B32, ASYNCcnt).
// Probe result: builtin takes (addrspace(1) int*, addrspace(3) int*, imm, imm).
// ---------------------------------------------------------------------------
#if __has_builtin(__builtin_amdgcn_global_load_async_to_lds_b32)
#define HAVE_ASYNC_LDS 1
typedef __attribute__((address_space(1))) int* gas_i32;
typedef __attribute__((address_space(3))) int* las_i32;
__device__ __forceinline__ void async_copy_b32(const unsigned int* g,
                                               unsigned int* l) {
  __builtin_amdgcn_global_load_async_to_lds_b32((gas_i32)g, (las_i32)l, 0, 0);
}
#else
#define HAVE_ASYNC_LDS 0
__device__ __forceinline__ void async_copy_b32(const unsigned int* g,
                                               unsigned int* l) {
  *l = *g;
}
#endif

template <int N>
__device__ __forceinline__ void wait_async_le() {
#if __has_builtin(__builtin_amdgcn_s_wait_asynccnt)
  __builtin_amdgcn_s_wait_asynccnt(N);
#else
  asm volatile("s_wait_asynccnt %0" ::"i"(N) : "memory");
#endif
}

// round-to-nearest-even f32 -> bf16, packed pair (lo -> [15:0], hi -> [31:16])
__device__ __forceinline__ unsigned int pack2_bf16(float lo, float hi) {
  unsigned int ul = __float_as_uint(lo);
  unsigned int uh = __float_as_uint(hi);
  ul += 0x7FFFu + ((ul >> 16) & 1u);
  uh += 0x7FFFu + ((uh >> 16) & 1u);
  return (ul >> 16) | (uh & 0xFFFF0000u);
}

// ---------------------------------------------------------------------------
// Pass 1a: channel-pair packing of in1 (unpadded).
// ---------------------------------------------------------------------------
__global__ __launch_bounds__(256)
void pack_pairs_kernel(const float* __restrict__ src,
                       unsigned int* __restrict__ dst, int total) {
  const int idx = blockIdx.x * blockDim.x + threadIdx.x;
  if (idx >= total) return;
  const int s  = idx % HW;
  const int bc = idx / HW;             // b*128 + cp
  const size_t lo = ((size_t)((bc >> 7) * NC + 2 * (bc & 127))) * HW + s;
  dst[idx] = pack2_bf16(src[lo], src[lo + HW]);
}

// ---------------------------------------------------------------------------
// Pass 1b: channel-pair packing of in2 into zero-padded [104][104] planes.
// ---------------------------------------------------------------------------
__global__ __launch_bounds__(256)
void pack_pairs_pad_kernel(const float* __restrict__ src,
                           unsigned int* __restrict__ dst, int total) {
  const int idx = blockIdx.x * blockDim.x + threadIdx.x;
  if (idx >= total) return;
  const int s  = idx % PHW;
  const int bc = idx / PHW;            // b*128 + cp
  const int i  = s / PW - MD;          // source row
  const int j  = s % PW - MD;          // source col
  unsigned int w = 0u;
  if ((unsigned)i < (unsigned)NH && (unsigned)j < (unsigned)NW) {
    const size_t lo = ((size_t)((bc >> 7) * NC + 2 * (bc & 127))) * HW + i * NW + j;
    w = pack2_bf16(src[lo], src[lo + HW]);
  }
  dst[idx] = w;
}

// ---------------------------------------------------------------------------
// Pass 2: WMMA correlation on packed bf16 pairs.
// ---------------------------------------------------------------------------
__global__ __launch_bounds__(NTHREADS)
void corr_wmma_main(const unsigned int* __restrict__ pk1,
                    const unsigned int* __restrict__ pk2,
                    float* __restrict__ out) {
  __shared__ __align__(16) unsigned int sA[2][ASZ];   // [buf][ti][m][kp]
  __shared__ __align__(16) unsigned int sB[2][BSZ];   // [buf][ri][a][kp]

  const int jt = blockIdx.x;                     // 0..5
  const int it = blockIdx.y;                     // 0..23
  const int b  = blockIdx.z;                     // 0..15
  const int j0 = jt * TJ;
  const int i0 = it * TI;

  const int tid  = threadIdx.x;
  const int wave = tid >> 5;                     // 0..8  (di = wave - 4)
  const int lane = tid & 31;
  const int h    = lane >> 4;                    // half-wave select
  const int m16  = lane & 15;

  // --- per-thread fill assignments (strength-reduced, constant strides) ----
  // A fill: waves 0..7 (tid<256), fixed (kp, m), 4 copies over ti.
  const int akp = (tid >> 4) & 15;
  const int am  = tid & 15;
  const unsigned int* gA0 =
      pk1 + (size_t)(b * PKC + akp) * HW + i0 * NW + (j0 + am);
  unsigned int* lA0 = &sA[0][am * KPP + akp];
  // B fill: all 288 threads, fixed (ri, a), 16 copies over kp.
  const int bri = tid / BCOL;                    // 0..11
  const int ba  = tid - bri * BCOL;              // 0..23
  const unsigned int* gB0 =
      pk2 + (size_t)(b * PKC) * PHW + (i0 + bri) * PW + (j0 + ba);
  unsigned int* lB0 = &sB[0][(bri * BCOL + ba) * KPP];

  // issue fills for one chunk into buffer `buf`
  auto issue_fill = [&](int kc, int buf) {
    if (tid < 256) {                             // wave-uniform branch
      const unsigned int* g = gA0 + (size_t)kc * (KP * HW);
      unsigned int* l = lA0 + buf * ASZ;
#pragma unroll
      for (int ti = 0; ti < TI; ++ti)
        async_copy_b32(g + ti * NW, l + ti * (TJ * KPP));
    }
    {
      const unsigned int* g = gB0 + (size_t)kc * (KP * PHW);
      unsigned int* l = lB0 + buf * BSZ;
#pragma unroll
      for (int kp = 0; kp < KP; ++kp)
        async_copy_b32(g + (size_t)kp * PHW, l + kp);
    }
  };

  v8f acc[TI][2] = {};

  issue_fill(0, 0);
  for (int kc = 0; kc < NKC; ++kc) {
    // Prefetch next chunk into the other buffer (its last readers finished
    // before the trailing barrier of iteration kc-1), then wait until only
    // the prefetch remains outstanding: async loads complete in order, so
    // chunk kc's copies have landed.  Per-wave counts: waves 0-7 issue
    // 4 (A) + 16 (B) = 20 copies, wave 8 issues 16.
    if (kc + 1 < NKC) {
      issue_fill(kc + 1, (kc + 1) & 1);
      if (wave < 8) wait_async_le<20>(); else wait_async_le<16>();
    } else {
      wait_async_le<0>();
    }
    __syncthreads();

    const unsigned int* sAc = sA[kc & 1];
    const unsigned int* sBc = sB[kc & 1];

    // ---- fragments (contiguous k = 16h + s in both A and B) + 8 WMMAs per wave per chunk ------
#pragma unroll
    for (int ti = 0; ti < TI; ++ti) {
      // A 16x32 bf16: lane needs k-pairs {h*8 .. h*8+7} @ m16
      const v4u* rowA = (const v4u*)&sAc[(ti * TJ + m16) * KPP];
      const v8u au = __builtin_bit_cast(
          v8u, __builtin_shufflevector(rowA[2 * h], rowA[2 * h + 1],
                                       0, 1, 2, 3, 4, 5, 6, 7));
      // B 32x16 bf16: lane needs k-pairs {h*8 .. h*8+7} @ column a
      const v4u* rowB1 = (const v4u*)&sBc[((wave + ti) * BCOL + m16) * KPP];
      const v4u* rowB2 = (const v4u*)&sBc[((wave + ti) * BCOL + m16 + 8) * KPP];
      const v8u b1u = __builtin_bit_cast(
          v8u, __builtin_shufflevector(rowB1[2 * h], rowB1[2 * h + 1],
                                       0, 1, 2, 3, 4, 5, 6, 7));
      const v8u b2u = __builtin_bit_cast(
          v8u, __builtin_shufflevector(rowB2[2 * h], rowB2[2 * h + 1],
                                       0, 1, 2, 3, 4, 5, 6, 7));
      const v16bf af  = __builtin_bit_cast(v16bf, au);
      const v16bf bf1 = __builtin_bit_cast(v16bf, b1u);
      const v16bf bf2 = __builtin_bit_cast(v16bf, b2u);
      acc[ti][0] = __builtin_amdgcn_wmma_f32_16x16x32_bf16(
          false, af, false, bf1, (short)0, acc[ti][0], false, false);
      acc[ti][1] = __builtin_amdgcn_wmma_f32_16x16x32_bf16(
          false, af, false, bf2, (short)0, acc[ti][1], false, false);
    }
    __syncthreads();
  }

  // ---- band extraction + scatter stores -----------------------------------
  // C/D layout: VGPR r, lane l -> M = r + 8*(l/16), N = l&15.
  // out element (m, dj) lives at band column a = m + dj + 4:
  // tile1 (a = N) when a <= 15, tile2 (a = N+8) when a >= 16.
  const float inv_c = 1.0f / (float)NC;
  const size_t outb = (size_t)b * 81 * HW;
#pragma unroll
  for (int ti = 0; ti < TI; ++ti) {
    const int i = i0 + ti;
#pragma unroll
    for (int r = 0; r < 8; ++r) {
      const int m = r + 8 * h;
      const int dj1 = m16 - m - MD;
      if (dj1 >= -MD && dj1 <= MD) {
        const int d = wave * ND + (dj1 + MD);
        out[outb + ((size_t)d * NH + i) * NW + (j0 + m)] = acc[ti][0][r] * inv_c;
      }
      const int dj2 = m16 + 8 - m - MD;
      if (m16 >= 8 && dj2 >= -MD && dj2 <= MD) {
        const int d = wave * ND + (dj2 + MD);
        out[outb + ((size_t)d * NH + i) * NW + (j0 + m)] = acc[ti][1][r] * inv_c;
      }
    }
  }
}

// ---------------------------------------------------------------------------
// Fallback: single-pass kernel (in-kernel conversion) if d_ws is too small.
// ---------------------------------------------------------------------------
__global__ __launch_bounds__(NTHREADS)
void corr_wmma_fallback(const float* __restrict__ in1,
                        const float* __restrict__ in2,
                        float* __restrict__ out) {
  __shared__ unsigned int sA[KP * TJ];
  __shared__ unsigned int sB[ND * KP * BCOL];

  const int jt = blockIdx.x;
  const int i  = blockIdx.y;
  const int b  = blockIdx.z;
  const int j0 = jt * TJ;

  const int tid  = threadIdx.x;
  const int wave = tid >> 5;
  const int lane = tid & 31;
  const int h    = lane >> 4;
  const int m16  = lane & 15;

  v8f acc1 = {};
  v8f acc2 = {};

  for (int kc = 0; kc < NKC; ++kc) {
    if (tid < KP * TJ) {
      const int kp = tid >> 4;
      const int m  = tid & 15;
      const int c  = kc * KC + kp * 2;
      const float* p = in1 + (((size_t)(b * NC + c) * NH + i) * NW + (j0 + m));
      sA[tid] = pack2_bf16(p[0], p[HW]);
    }
    for (int idx = tid; idx < ND * KP * BCOL; idx += NTHREADS) {
      const int di  = idx / (KP * BCOL);
      const int rem = idx - di * (KP * BCOL);
      const int kp  = rem / BCOL;
      const int a   = rem - kp * BCOL;
      const int row = i + di - MD;
      const int col = j0 - MD + a;
      float lo = 0.0f, hi = 0.0f;
      if ((unsigned)row < (unsigned)NH && (unsigned)col < (unsigned)NW) {
        const int c = kc * KC + kp * 2;
        const float* p = in2 + (((size_t)(b * NC + c) * NH + row) * NW + col);
        lo = p[0];
        hi = p[HW];
      }
      sB[idx] = pack2_bf16(lo, hi);
    }
    __syncthreads();

    v8u au, b1u, b2u;
#pragma unroll
    for (int v = 0; v < 8; ++v) {
      const int kpA = h * 8 + v;
      au[v] = sA[kpA * TJ + m16];
      const int kpB = h * 8 + v;
      const int bb  = (wave * KP + kpB) * BCOL;
      b1u[v] = sB[bb + m16];
      b2u[v] = sB[bb + m16 + 8];
    }
    const v16bf af  = __builtin_bit_cast(v16bf, au);
    const v16bf bf1 = __builtin_bit_cast(v16bf, b1u);
    const v16bf bf2 = __builtin_bit_cast(v16bf, b2u);
    acc1 = __builtin_amdgcn_wmma_f32_16x16x32_bf16(false, af, false, bf1,
                                                   (short)0, acc1, false, false);
    acc2 = __builtin_amdgcn_wmma_f32_16x16x32_bf16(false, af, false, bf2,
                                                   (short)0, acc2, false, false);
    __syncthreads();
  }

  const float inv_c = 1.0f / (float)NC;
  const size_t outb = (size_t)b * 81 * HW;
#pragma unroll
  for (int r = 0; r < 8; ++r) {
    const int m = r + 8 * h;
    const int dj1 = m16 - m - MD;
    if (dj1 >= -MD && dj1 <= MD) {
      const int d = wave * ND + (dj1 + MD);
      out[outb + ((size_t)d * NH + i) * NW + (j0 + m)] = acc1[r] * inv_c;
    }
    const int dj2 = m16 + 8 - m - MD;
    if (m16 >= 8 && dj2 >= -MD && dj2 <= MD) {
      const int d = wave * ND + (dj2 + MD);
      out[outb + ((size_t)d * NH + i) * NW + (j0 + m)] = acc2[r] * inv_c;
    }
  }
}

extern "C" void kernel_launch(void* const* d_in, const int* in_sizes, int n_in,
                              void* d_out, int out_size, void* d_ws, size_t ws_size,
                              hipStream_t stream) {
  (void)in_sizes; (void)n_in; (void)out_size;
  const float* in1 = (const float*)d_in[0];
  const float* in2 = (const float*)d_in[1];
  float* out = (float*)d_out;

  const size_t pk1_elems = (size_t)NB * PKC * HW;    // 18,874,368 dwords
  const size_t pk2_elems = (size_t)NB * PKC * PHW;   // 22,151,168 dwords
  const size_t need = (pk1_elems + pk2_elems) * sizeof(unsigned int);  // ~164 MB

  if (ws_size >= need) {
    unsigned int* pk1 = (unsigned int*)d_ws;
    unsigned int* pk2 = pk1 + pk1_elems;
    pack_pairs_kernel<<<((int)pk1_elems + 255) / 256, 256, 0, stream>>>(
        in1, pk1, (int)pk1_elems);
    pack_pairs_pad_kernel<<<((int)pk2_elems + 255) / 256, 256, 0, stream>>>(
        in2, pk2, (int)pk2_elems);
    corr_wmma_main<<<dim3(NW / TJ, NH / TI, NB), NTHREADS, 0, stream>>>(pk1, pk2, out);
  } else {
    corr_wmma_fallback<<<dim3(NW / TJ, NH, NB), NTHREADS, 0, stream>>>(in1, in2, out);
  }
}
